// Attention_163208757610
// MI455X (gfx1250) — matrix operand order repaired and checked by reading, not compiled, not run on hardware
//
#include <hip/hip_runtime.h>
#include <hip/hip_bf16.h>

typedef __attribute__((ext_vector_type(16))) _Float16 v16h;
typedef __attribute__((ext_vector_type(8)))  _Float16 v8h;
typedef __attribute__((ext_vector_type(8)))  float    v8f;

#define HEADS 16
#define DH 64
#define NSEQ 2048
#define MSEQ 2048
#define DMODEL 1024
#define INNER 1024
#define BATCH 2
#define ROWS (BATCH*NSEQ)   // 4096

// ---------------- WMMA helpers (layouts per CDNA5 ISA 7.12.2) ----------------

__device__ __forceinline__ v8f wmma_f16(v16h a, v16h b, v8f c) {
  // v_wmma_f32_16x16x32_f16
  return __builtin_amdgcn_wmma_f32_16x16x32_f16(false, a, false, b, (short)0, c,
                                                false, false);
}

// A fragment: 16x32 f16, row = lane%16, halves 0..7 -> K=base..base+7,
// halves 8..15 -> K=base+16..base+23, base = (lane<16 ? 0 : 8).
__device__ __forceinline__ v16h load_afrag(const _Float16* A, int lda, int k0) {
  const int lane = threadIdx.x & 31;
  const _Float16* p = A + (lane & 15) * lda + k0 + ((lane < 16) ? 0 : 8);
  v8h lo = *(const v8h*)p;
  v8h hi = *(const v8h*)(p + 16);
  v16h r;
#pragma unroll
  for (int i = 0; i < 8; ++i) { r[i] = lo[i]; r[i + 8] = hi[i]; }
  return r;
}

// B fragment: 32x16 f16, col = lane%16, halves 0..7 -> K=base..base+7,
// halves 8..15 -> K=base+16..base+23, base = (lane<16 ? 0 : 8).  Bt laid out so K is contiguous per column.
__device__ __forceinline__ v16h load_bfrag(const _Float16* Bt, int ldb, int k0) {
  const int lane = threadIdx.x & 31;
  const _Float16* p = Bt + (lane & 15) * ldb + k0 + ((lane < 16) ? 0 : 8);
  v8h lo = *(const v8h*)p;
  v8h hi = *(const v8h*)(p + 16);
  v16h r;
#pragma unroll
  for (int i = 0; i < 8; ++i) { r[i] = lo[i]; r[i + 8] = hi[i]; }
  return r;
}

// Ping-pong double-buffered 16x64 GEMM tile: acc[0..3] += A(16x1024)*B(1024x64).
// k-loop unrolled by 2 so the two fragment sets alternate roles -> no register
// copies between buffers (the round-3 build spent ~40 v_mov/iter on rotation).
__device__ __forceinline__ void gemm_tile_16x64(const _Float16* __restrict__ Arow,
                                                const _Float16* __restrict__ Bbase,
                                                v8f acc[4]) {
  v16h a0, a1, b0[4], b1[4];
  a0 = load_afrag(Arow, DMODEL, 0);
#pragma unroll
  for (int t = 0; t < 4; ++t)
    b0[t] = load_bfrag(Bbase + (size_t)t * 16 * DMODEL, DMODEL, 0);
  int kk = 0;
  for (; kk < DMODEL - 64; kk += 64) {
    a1 = load_afrag(Arow, DMODEL, kk + 32);
#pragma unroll
    for (int t = 0; t < 4; ++t)
      b1[t] = load_bfrag(Bbase + (size_t)t * 16 * DMODEL, DMODEL, kk + 32);
#pragma unroll
    for (int t = 0; t < 4; ++t) acc[t] = wmma_f16(a0, b0[t], acc[t]);
    a0 = load_afrag(Arow, DMODEL, kk + 64);
#pragma unroll
    for (int t = 0; t < 4; ++t)
      b0[t] = load_bfrag(Bbase + (size_t)t * 16 * DMODEL, DMODEL, kk + 64);
#pragma unroll
    for (int t = 0; t < 4; ++t) acc[t] = wmma_f16(a1, b1[t], acc[t]);
  }
  // tail: a0/b0 hold k = DMODEL-64; load final k-step and drain
  a1 = load_afrag(Arow, DMODEL, DMODEL - 32);
#pragma unroll
  for (int t = 0; t < 4; ++t)
    b1[t] = load_bfrag(Bbase + (size_t)t * 16 * DMODEL, DMODEL, DMODEL - 32);
#pragma unroll
  for (int t = 0; t < 4; ++t) acc[t] = wmma_f16(a0, b0[t], acc[t]);
#pragma unroll
  for (int t = 0; t < 4; ++t) acc[t] = wmma_f16(a1, b1[t], acc[t]);
}

// ---------------- async global->LDS staging (CDNA5, ASYNCcnt) ----------------
// Copies a 4KB tile (32 rows x 64 f16) with 8 x global_load_async_to_lds_b128.
__device__ __forceinline__ void async_copy_4k(const _Float16* gsrc, unsigned lds_base) {
  const int lane = threadIdx.x & 31;
  unsigned loff = lds_base + (unsigned)lane * 16u;
  unsigned long long gaddr = (unsigned long long)(uintptr_t)gsrc + (unsigned)lane * 16u;
#pragma unroll
  for (int i = 0; i < 8; ++i) {
    asm volatile("global_load_async_to_lds_b128 %0, %1, off"
                 :: "v"(loff), "v"(gaddr) : "memory");
    loff += 512u;
    gaddr += 512u;
  }
}
__device__ __forceinline__ void wait_async_le(int prev_in_flight) {
  if (prev_in_flight)
    asm volatile("s_wait_asynccnt 0x8" ::: "memory");
  else
    asm volatile("s_wait_asynccnt 0x0" ::: "memory");
}

// ---------------- conversion kernels ----------------

__global__ void cvt_f32_f16(const float* __restrict__ src,
                            _Float16* __restrict__ dst, int n) {
  int i = blockIdx.x * blockDim.x + threadIdx.x;
  if (i < n) dst[i] = (_Float16)src[i];
}

// src[K][Ncols] (f32, row-major) -> dst[Ncols][K] (f16)
__global__ void cvt_transpose_f16(const float* __restrict__ src,
                                  _Float16* __restrict__ dst, int K, int Ncols) {
  int i = blockIdx.x * blockDim.x + threadIdx.x;
  if (i < K * Ncols) {
    int k = i / Ncols, n = i - k * Ncols;
    dst[n * K + k] = (_Float16)src[i];
  }
}

// ---------------- projection GEMMs ----------------
// block = 128 threads (4 waves); wave computes a 16x64 tile (4 WMMA accums).
// __launch_bounds__(128,1): relax occupancy so the register allocator keeps the
// double-buffered fragments in VGPRs (no scratch spills in the inner loop).

__global__ void __launch_bounds__(128, 1)
gemm_q(const _Float16* __restrict__ X,   // [4096][1024]
       const _Float16* __restrict__ WqT, // [1024][1024]
       _Float16* __restrict__ Q) {       // [b][h][N][64], scaled
  const int wave = threadIdx.x >> 5, lane = threadIdx.x & 31;
  const int m0 = blockIdx.x * 16;
  const int n0 = (blockIdx.y * 4 + wave) * 64;
  v8f acc[4] = {{}, {}, {}, {}};
  gemm_tile_16x64(X + (size_t)m0 * DMODEL, WqT + (size_t)n0 * DMODEL, acc);
  const int col = lane & 15, hs = (lane >> 4) * 8;
#pragma unroll
  for (int t = 0; t < 4; ++t)
#pragma unroll
    for (int r = 0; r < 8; ++r) {
      int m = m0 + r + hs, n = n0 + t * 16 + col;
      int b = m >> 11, i = m & 2047, h = n >> 6, d = n & 63;
      Q[(((size_t)(b * HEADS + h) * NSEQ) + i) * DH + d] =
          (_Float16)(acc[t][r] * 0.125f);  // fold SCALE = 1/sqrt(64)
    }
}

__global__ void __launch_bounds__(128, 1)
gemm_kv(const _Float16* __restrict__ C,    // [4096][1024]
        const _Float16* __restrict__ WkvT, // [2048][1024]
        _Float16* __restrict__ K,          // [b][h][M][64]
        _Float16* __restrict__ VT) {       // [b][h][64][M]
  const int wave = threadIdx.x >> 5, lane = threadIdx.x & 31;
  const int m0 = blockIdx.x * 16;
  const int n0 = (blockIdx.y * 4 + wave) * 64;
  v8f acc[4] = {{}, {}, {}, {}};
  gemm_tile_16x64(C + (size_t)m0 * DMODEL, WkvT + (size_t)n0 * DMODEL, acc);
  const int col = lane & 15, hs = (lane >> 4) * 8;
#pragma unroll
  for (int t = 0; t < 4; ++t)
#pragma unroll
    for (int r = 0; r < 8; ++r) {
      int m = m0 + r + hs, n = n0 + t * 16 + col;
      int b = m >> 11, j = m & 2047;
      _Float16 v = (_Float16)acc[t][r];
      if (n < INNER) {
        int h = n >> 6, d = n & 63;
        K[(((size_t)(b * HEADS + h) * MSEQ) + j) * DH + d] = v;
      } else {
        int n2 = n - INNER, h = n2 >> 6, d = n2 & 63;
        VT[(((size_t)(b * HEADS + h) * DH) + d) * MSEQ + j] = v;
      }
    }
}

// ---------------- fused flash attention ----------------
// grid (N/16/NW, B*H), block 128 (4 waves); each wave owns a 16-row q tile.
// K tiles are staged into LDS with async global->LDS DMA, double-buffered.

#define NW 4

__global__ void __launch_bounds__(128, 1)
attn_flash(const _Float16* __restrict__ Q,   // [b][h][N][64]
           const _Float16* __restrict__ K,   // [b][h][M][64]
           const _Float16* __restrict__ VT,  // [b][h][64][M]
           const unsigned char* __restrict__ mask, // [b][M]
           _Float16* __restrict__ O) {       // [b][N][1024]
  __shared__ __attribute__((aligned(16))) _Float16 pbuf[NW][16 * 32];
  __shared__ __attribute__((aligned(16))) _Float16 kbuf[NW][2][32 * DH];
  const int wave = threadIdx.x >> 5, lane = threadIdx.x & 31;
  const int bh = blockIdx.y, b = bh >> 4, h = bh & 15;
  const int i0 = (blockIdx.x * NW + wave) * 16;
  const int col = lane & 15, hs = (lane >> 4) * 8;

  const _Float16* Qb = Q + ((size_t)bh * NSEQ + i0) * DH;
  const _Float16* Kb = K + (size_t)bh * MSEQ * DH;
  const _Float16* Vb = VT + (size_t)bh * DH * MSEQ;
  const unsigned char* mb = mask + (size_t)b * MSEQ;

  // LDS byte offsets of this wave's K staging buffers (addr[31:0] of generic)
  unsigned kbase[2];
  kbase[0] = (unsigned)(uintptr_t)&kbuf[wave][0][0];
  kbase[1] = (unsigned)(uintptr_t)&kbuf[wave][1][0];

  v16h qf[2];
  qf[0] = load_afrag(Qb, DH, 0);
  qf[1] = load_afrag(Qb, DH, 32);

  v8f acc[4] = {{}, {}, {}, {}};
  float mrow[8], lrow[8];
#pragma unroll
  for (int r = 0; r < 8; ++r) { mrow[r] = -3.0e38f; lrow[r] = 0.0f; }

  // stage first K tile
  async_copy_4k(Kb, kbase[0]);
  int cur = 0;

  for (int j0 = 0; j0 < MSEQ; j0 += 32) {
    const int has_next = (j0 + 32 < MSEQ);
    if (has_next) {
      // kick off DMA for the next 32xDH K tile into the other buffer
      async_copy_4k(Kb + (size_t)(j0 + 32) * DH, kbase[cur ^ 1]);
      __builtin_prefetch(Vb + (size_t)(j0 + 32), 0, 3);
    }
    wait_async_le(has_next);  // current tile resident in LDS
    const _Float16* Kt = &kbuf[wave][cur][0];  // [32][DH] row-major in LDS

    // S tile 16x32 via 4 WMMAs (2 j-subtiles x 2 k-steps over d=64)
    v8f s[2];
#pragma unroll
    for (int t = 0; t < 2; ++t) {
      v8f sf = {};
#pragma unroll
      for (int ks = 0; ks < 2; ++ks) {
        v16h kf = load_bfrag(Kt + (size_t)(t * 16) * DH, DH, ks * 32);
        sf = wmma_f16(qf[ks], kf, sf);
      }
      unsigned char mv = mb[j0 + t * 16 + col];
      if (!mv) {
#pragma unroll
        for (int r = 0; r < 8; ++r) sf[r] = -3.0e38f;
      }
      s[t] = sf;
    }

    // online softmax: rows live across the 16-lane half; reduce with shfl_xor
    float nm[8], alpha[8];
#pragma unroll
    for (int r = 0; r < 8; ++r) {
      float x = fmaxf(s[0][r], s[1][r]);
#pragma unroll
      for (int off = 1; off < 16; off <<= 1)
        x = fmaxf(x, __shfl_xor(x, off, 32));
      nm[r] = fmaxf(mrow[r], x);
      alpha[r] = __expf(mrow[r] - nm[r]);
    }
#pragma unroll
    for (int r = 0; r < 8; ++r) {
      float p0 = __expf(s[0][r] - nm[r]);
      float p1 = __expf(s[1][r] - nm[r]);
      s[0][r] = p0; s[1][r] = p1;
      float rs = p0 + p1;
#pragma unroll
      for (int off = 1; off < 16; off <<= 1)
        rs += __shfl_xor(rs, off, 32);
      lrow[r] = lrow[r] * alpha[r] + rs;
      mrow[r] = nm[r];
    }
#pragma unroll
    for (int dt = 0; dt < 4; ++dt)
#pragma unroll
      for (int r = 0; r < 8; ++r) acc[dt][r] *= alpha[r];

    // Re-layout P (C/D frag -> A frag) through LDS
    __syncthreads();
#pragma unroll
    for (int t = 0; t < 2; ++t)
#pragma unroll
      for (int r = 0; r < 8; ++r)
        pbuf[wave][(r + hs) * 32 + t * 16 + col] = (_Float16)s[t][r];
    __syncthreads();
    v16h pf = load_afrag(&pbuf[wave][0], 32, 0);

    // acc(16x64) += P(16x32) @ V(32x64); VT gives contiguous B-frag loads
#pragma unroll
    for (int dt = 0; dt < 4; ++dt) {
      v16h vf = load_bfrag(Vb + (size_t)(dt * 16) * MSEQ, MSEQ, j0);
      acc[dt] = wmma_f16(pf, vf, acc[dt]);
    }
    cur ^= 1;
  }

  // epilogue: normalize and scatter to O[b][i][h*64+d]
#pragma unroll
  for (int dt = 0; dt < 4; ++dt)
#pragma unroll
    for (int r = 0; r < 8; ++r) {
      int i = i0 + r + hs;
      int d = h * DH + dt * 16 + col;
      float l = lrow[r];
      l = (l != 0.0f) ? l : 1.0f;
      O[((size_t)b * NSEQ + i) * INNER + d] = (_Float16)(acc[dt][r] / l);
    }
}

// ---------------- output projection ----------------

__global__ void __launch_bounds__(128, 1)
gemm_out(const _Float16* __restrict__ O,   // [4096][1024]
         const _Float16* __restrict__ WoT, // [1024][1024]
         const float* __restrict__ bo,     // [1024]
         float* __restrict__ out) {        // [4096][1024] f32
  const int wave = threadIdx.x >> 5, lane = threadIdx.x & 31;
  const int m0 = blockIdx.x * 16;
  const int n0 = (blockIdx.y * 4 + wave) * 64;
  v8f acc[4] = {{}, {}, {}, {}};
  gemm_tile_16x64(O + (size_t)m0 * INNER, WoT + (size_t)n0 * INNER, acc);
  const int col = lane & 15, hs = (lane >> 4) * 8;
#pragma unroll
  for (int t = 0; t < 4; ++t)
#pragma unroll
    for (int r = 0; r < 8; ++r) {
      int m = m0 + r + hs, n = n0 + t * 16 + col;
      out[(size_t)m * DMODEL + n] = acc[t][r] + bo[n];
    }
}

// ---------------- host launch ----------------

extern "C" void kernel_launch(void* const* d_in, const int* in_sizes, int n_in,
                              void* d_out, int out_size, void* d_ws, size_t ws_size,
                              hipStream_t stream) {
  const float* x       = (const float*)d_in[0];
  const float* context = (const float*)d_in[1];
  const unsigned char* mask = (const unsigned char*)d_in[2]; // jax bool = 1 byte
  const float* Wq  = (const float*)d_in[3];
  const float* Wkv = (const float*)d_in[4];
  const float* Wo  = (const float*)d_in[5];
  const float* bo  = (const float*)d_in[6];
  float* out = (float*)d_out;

  char* ws = (char*)d_ws;
  // byte offsets (all 256B aligned)
  _Float16* X16  = (_Float16*)(ws + 0);          //  8 MiB: [4096][1024]
  _Float16* C16  = (_Float16*)(ws + 8388608);    //  8 MiB: [4096][1024]
  _Float16* WQT  = (_Float16*)(ws + 16777216);   //  2 MiB: [1024][1024]
  _Float16* WKVT = (_Float16*)(ws + 18874368);   //  4 MiB: [2048][1024]
  _Float16* WOT  = (_Float16*)(ws + 23068672);   //  2 MiB: [1024][1024]
  _Float16* Q16  = (_Float16*)(ws + 25165824);   //  8 MiB: [b][h][N][64]
  _Float16* K16  = (_Float16*)(ws + 33554432);   //  8 MiB: [b][h][M][64]
  _Float16* VT16 = (_Float16*)(ws + 41943040);   //  8 MiB: [b][h][64][M]
  _Float16* O16  = (_Float16*)(ws + 50331648);   //  8 MiB: [4096][1024]

  const int nX = ROWS * DMODEL;           // 4,194,304
  const int nWq = DMODEL * INNER;         // 1,048,576
  const int nWkv = DMODEL * 2 * INNER;    // 2,097,152

  cvt_f32_f16<<<(nX + 255) / 256, 256, 0, stream>>>(x, X16, nX);
  cvt_f32_f16<<<(nX + 255) / 256, 256, 0, stream>>>(context, C16, nX);
  cvt_transpose_f16<<<(nWq + 255) / 256, 256, 0, stream>>>(Wq, WQT, DMODEL, INNER);
  cvt_transpose_f16<<<(nWkv + 255) / 256, 256, 0, stream>>>(Wkv, WKVT, DMODEL, 2 * INNER);
  cvt_transpose_f16<<<(nWq + 255) / 256, 256, 0, stream>>>(Wo, WOT, INNER, DMODEL);

  gemm_q <<<dim3(ROWS / 16, INNER / 256),      128, 0, stream>>>(X16, WQT, Q16);
  gemm_kv<<<dim3(ROWS / 16, 2 * INNER / 256),  128, 0, stream>>>(C16, WKVT, K16, VT16);
  attn_flash<<<dim3(NSEQ / 16 / NW, BATCH * HEADS), 32 * NW, 0, stream>>>(
      Q16, K16, VT16, mask, O16);
  gemm_out<<<dim3(ROWS / 16, DMODEL / 256), 128, 0, stream>>>(O16, WOT, bo, out);
}
